// MatrixLayer_9483287789979
// MI455X (gfx1250) — compile-verified, matrix operand order checked by reading, not run on hardware
//
#include <hip/hip_runtime.h>
#include <hip/hip_bf16.h>

// ---------------------------------------------------------------------------
// MI455X (gfx1250) implementation.
//  GEMMs via v_wmma_f32_16x16x32_bf16 (wave32, 16x16 tiles, f32 accum).
//  Prefix 2x2-matmul scan via 3-phase hierarchical scan (512 WGs).
//  GEMM2 A-tiles staged with global_load_async_to_lds_b128 (ASYNCcnt).
//  bf16 packing via hardware v_cvt_pk_bf16_f32 (__builtin_convertvector).
// ---------------------------------------------------------------------------

typedef __bf16 v16bf  __attribute__((ext_vector_type(16)));
typedef __bf16 bf16x2 __attribute__((ext_vector_type(2)));
typedef float  v8f    __attribute__((ext_vector_type(8)));
typedef float  f32x2  __attribute__((ext_vector_type(2)));

#define B_     4
#define T_     2048
#define D_     1024
#define K_     256          // k
#define RN     8192         // B*T
#define FOURK  1024         // 4*k
#define TWOK   512          // 2*k
#define NT     16           // number of T-chunks in the scan
#define CHUNK  128          // timesteps per chunk (NT*CHUNK == T_)
#define SUB    16           // timesteps per lane sub-chunk (8 waves * 16 = 128)
#define EPS_RMS 1.1920929e-07f

// hardware packed f32->bf16 conversion (v_cvt_pk_bf16_f32)
static __device__ __forceinline__ unsigned pack2(float lo, float hi) {
  f32x2 f = {lo, hi};
  bf16x2 b = __builtin_convertvector(f, bf16x2);
  return __builtin_bit_cast(unsigned, b);
}
static __device__ __forceinline__ unsigned short f2bf(float f) {
  __bf16 b = (__bf16)f;
  return __builtin_bit_cast(unsigned short, b);
}
static __device__ __forceinline__ float bf2f(unsigned short s) {
  return __uint_as_float(((unsigned)s) << 16);
}

struct m22 { float a, b, c, d; };        // row-major [[a,b],[c,d]]
static __device__ __forceinline__ m22 mmul(const m22 X, const m22 Y) {
  m22 r;
  r.a = X.a * Y.a + X.b * Y.c;
  r.b = X.a * Y.b + X.b * Y.d;
  r.c = X.c * Y.a + X.d * Y.c;
  r.d = X.c * Y.b + X.d * Y.d;
  return r;
}

// ---------------------------------------------------------------------------
// Kernel 0: per-row 1/rms.  One wave per row of x; float4 (b128) loads.
// ---------------------------------------------------------------------------
__global__ __launch_bounds__(256)
void k_rrms(const float* __restrict__ x, float* __restrict__ rrms) {
  const int row  = blockIdx.x * 8 + (threadIdx.x >> 5);
  const int lane = threadIdx.x & 31;
  const float4* p = (const float4*)(x + (size_t)row * D_);
  float s = 0.f;
#pragma unroll
  for (int i = 0; i < D_ / 128; ++i) {       // 8 x float4 per lane
    float4 v = p[lane + 32 * i];
    s += v.x * v.x + v.y * v.y + v.z * v.z + v.w * v.w;
  }
#pragma unroll
  for (int m = 16; m; m >>= 1) s += __shfl_xor(s, m, 32);
  if (lane == 0) rrms[row] = rsqrtf(s * (1.0f / D_) + EPS_RMS);
}

// ---------------------------------------------------------------------------
// Kernel 1: M = rmsnorm(x) @ W_m^T, then per-2x2 Frobenius normalization.
// 64(M) x 128(N) tile per 256-thread WG; 8 waves in a 2x4 grid, each wave
// owns a 32x32 tile (2x2 WMMA accumulators).  K loop of 32 over D=1024.
// All staging is 16B/thread: float4 global loads -> packed uint4 LDS stores.
// ---------------------------------------------------------------------------
__global__ __launch_bounds__(256)
void k_gemm1(const float* __restrict__ x, const float* __restrict__ Wm,
             const float* __restrict__ rrms, unsigned short* __restrict__ Mout) {
  __shared__ __align__(32) unsigned short lsA[64 * 32];
  __shared__ __align__(32) unsigned short lsB[128 * 32];

  const int tid   = threadIdx.x;
  const int lane  = tid & 31;
  const int wid   = tid >> 5;
  const int waveM = wid & 1;          // 0..1
  const int waveN = wid >> 1;         // 0..3
  const int rbase = blockIdx.x * 64;
  const int nbase = blockIdx.y * 128;

  // staging coordinates (fixed per thread)
  const int ra  = tid >> 2;           // 0..63   A row
  const int ca  = (tid & 3) * 8;      // A col group (8 halfs)
  const int rb  = tid >> 1;           // 0..127  B row
  const int cb  = (tid & 1) * 16;     // B col group (16 halfs)
  const float scA = rrms[rbase + ra];

  v8f acc[2][2] = {};

  for (int kt = 0; kt < D_ / 32; ++kt) {
    const int kb = kt * 32;
    if (kt + 1 < D_ / 32) {
      __builtin_prefetch(&x[(size_t)(rbase + ra) * D_ + kb + 32 + ca], 0, 1);
      __builtin_prefetch(&Wm[(size_t)(nbase + rb) * D_ + kb + 32 + cb], 0, 1);
    }
    // stage A: 64x32 of rmsnorm(x) -> bf16 (2 x b128 loads, 1 x b128 store)
    {
      const float4* src = (const float4*)&x[(size_t)(rbase + ra) * D_ + kb + ca];
      float4 f0 = src[0], f1 = src[1];
      uint4 pk;
      pk.x = pack2(f0.x * scA, f0.y * scA);
      pk.y = pack2(f0.z * scA, f0.w * scA);
      pk.z = pack2(f1.x * scA, f1.y * scA);
      pk.w = pack2(f1.z * scA, f1.w * scA);
      *(uint4*)&lsA[ra * 32 + ca] = pk;
    }
    // stage B: 128x32 of W_m rows -> bf16 (4 x b128 loads, 2 x b128 stores)
    {
      const float4* src = (const float4*)&Wm[(size_t)(nbase + rb) * D_ + kb + cb];
      float4 f0 = src[0], f1 = src[1], f2 = src[2], f3 = src[3];
      uint4 p0, p1;
      p0.x = pack2(f0.x, f0.y); p0.y = pack2(f0.z, f0.w);
      p0.z = pack2(f1.x, f1.y); p0.w = pack2(f1.z, f1.w);
      p1.x = pack2(f2.x, f2.y); p1.y = pack2(f2.z, f2.w);
      p1.z = pack2(f3.x, f3.y); p1.w = pack2(f3.z, f3.w);
      *(uint4*)&lsB[rb * 32 + cb]     = p0;
      *(uint4*)&lsB[rb * 32 + cb + 8] = p1;
    }
    __syncthreads();

    const int klo = (lane >> 4) * 16;   // lanes 0-15: K 0..15, 16-31: K 16..31
    const int l15 = lane & 15;
    const int ar  = waveM * 32;
    const int bc  = waveN * 32;
    v16bf a0 = *(const v16bf*)&lsA[(ar + l15) * 32 + klo];
    v16bf a1 = *(const v16bf*)&lsA[(ar + 16 + l15) * 32 + klo];
    v16bf b0 = *(const v16bf*)&lsB[(bc + l15) * 32 + klo];
    v16bf b1 = *(const v16bf*)&lsB[(bc + 16 + l15) * 32 + klo];

    acc[0][0] = __builtin_amdgcn_wmma_f32_16x16x32_bf16(false, a0, false, b0, (short)0, acc[0][0], false, false);
    acc[0][1] = __builtin_amdgcn_wmma_f32_16x16x32_bf16(false, a0, false, b1, (short)0, acc[0][1], false, false);
    acc[1][0] = __builtin_amdgcn_wmma_f32_16x16x32_bf16(false, a1, false, b0, (short)0, acc[1][0], false, false);
    acc[1][1] = __builtin_amdgcn_wmma_f32_16x16x32_bf16(false, a1, false, b1, (short)0, acc[1][1], false, false);
    __syncthreads();
  }

  // Epilogue: 2x2 Frobenius normalization.  The 4 elements of a 2x2 block are
  // 4 adjacent lanes (cols 4q..4q+3) -> cross-lane sum via shfl_xor 1,2.
  const int hi8 = (lane >> 4) * 8;
#pragma unroll
  for (int mi = 0; mi < 2; ++mi)
#pragma unroll
    for (int ni = 0; ni < 2; ++ni) {
#pragma unroll
      for (int j = 0; j < 8; ++j) {
        float v = acc[mi][ni][j];
        float s = v * v;
        s += __shfl_xor(s, 1, 32);
        s += __shfl_xor(s, 2, 32);
        float inv = 1.0f / fmaxf(sqrtf(s), 1e-6f);
        int row = rbase + waveM * 32 + mi * 16 + hi8 + j;
        int col = nbase + waveN * 32 + ni * 16 + (lane & 15);
        Mout[(size_t)row * FOURK + col] = f2bf(v * inv);
      }
    }
}

// ---------------------------------------------------------------------------
// Scan phase A: per-chunk total products.
// ---------------------------------------------------------------------------
__global__ __launch_bounds__(256)
void k_scan_tot(const unsigned short* __restrict__ M, float* __restrict__ Ctot) {
  __shared__ float lds[8][32][4];
  const int bx    = blockIdx.x;
  const int chunk = bx & (NT - 1);
  const int kg    = (bx >> 4) & 7;
  const int b     = bx >> 7;
  const int lane  = threadIdx.x & 31;
  const int w     = threadIdx.x >> 5;
  const int chain = kg * 32 + lane;
  const int t0    = chunk * CHUNK + w * SUB;

  const unsigned short* mp = M + (((size_t)(b * T_ + t0)) * K_ + chain) * 4;
  m22 P = {1.f, 0.f, 0.f, 1.f};
#pragma unroll
  for (int j = 0; j < SUB; ++j) {
    uint2 u = *(const uint2*)mp;
    m22 m = {bf2f((unsigned short)(u.x & 0xFFFFu)), bf2f((unsigned short)(u.x >> 16)),
             bf2f((unsigned short)(u.y & 0xFFFFu)), bf2f((unsigned short)(u.y >> 16))};
    P = mmul(m, P);                       // later timestep on the left
    mp += (size_t)K_ * 4;
  }
  lds[w][lane][0] = P.a; lds[w][lane][1] = P.b;
  lds[w][lane][2] = P.c; lds[w][lane][3] = P.d;
  __syncthreads();
  if (w == 0) {
    m22 Tt = {lds[0][lane][0], lds[0][lane][1], lds[0][lane][2], lds[0][lane][3]};
#pragma unroll
    for (int ww = 1; ww < 8; ++ww) {
      m22 q = {lds[ww][lane][0], lds[ww][lane][1], lds[ww][lane][2], lds[ww][lane][3]};
      Tt = mmul(q, Tt);
    }
    float* o = Ctot + (((size_t)(b * NT + chunk)) * K_ + chain) * 4;
    o[0] = Tt.a; o[1] = Tt.b; o[2] = Tt.c; o[3] = Tt.d;
  }
}

// ---------------------------------------------------------------------------
// Scan phase B: exclusive scan of the 16 chunk totals per chain (1024 chains).
// ---------------------------------------------------------------------------
__global__ __launch_bounds__(256)
void k_scan_pre(const float* __restrict__ Ctot, float* __restrict__ Cpre) {
  const int gid   = blockIdx.x * 256 + threadIdx.x;   // 0..1023
  const int b     = gid >> 8;
  const int chain = gid & 255;
  m22 E = {1.f, 0.f, 0.f, 1.f};
#pragma unroll
  for (int c = 0; c < NT; ++c) {
    float* o = Cpre + (((size_t)(b * NT + c)) * K_ + chain) * 4;
    o[0] = E.a; o[1] = E.b; o[2] = E.c; o[3] = E.d;
    const float* p = Ctot + (((size_t)(b * NT + c)) * K_ + chain) * 4;
    m22 q = {p[0], p[1], p[2], p[3]};
    E = mmul(q, E);
  }
}

// ---------------------------------------------------------------------------
// Scan phase C: apply prefixes, compute h[t] = S[t] @ v, write bf16 h.
// ---------------------------------------------------------------------------
__global__ __launch_bounds__(256)
void k_scan_apply(const unsigned short* __restrict__ M,
                  const float* __restrict__ Cpre,
                  const float* __restrict__ vvec,
                  unsigned short* __restrict__ h) {
  __shared__ float lds[8][32][4];
  const int bx    = blockIdx.x;
  const int chunk = bx & (NT - 1);
  const int kg    = (bx >> 4) & 7;
  const int b     = bx >> 7;
  const int lane  = threadIdx.x & 31;
  const int w     = threadIdx.x >> 5;
  const int chain = kg * 32 + lane;
  const int t0    = chunk * CHUNK + w * SUB;

  const unsigned short* mp = M + (((size_t)(b * T_ + t0)) * K_ + chain) * 4;
  m22 mreg[SUB];
  m22 Q = {1.f, 0.f, 0.f, 1.f};
#pragma unroll
  for (int j = 0; j < SUB; ++j) {
    uint2 u = *(const uint2*)mp;
    mreg[j].a = bf2f((unsigned short)(u.x & 0xFFFFu));
    mreg[j].b = bf2f((unsigned short)(u.x >> 16));
    mreg[j].c = bf2f((unsigned short)(u.y & 0xFFFFu));
    mreg[j].d = bf2f((unsigned short)(u.y >> 16));
    Q = mmul(mreg[j], Q);
    mp += (size_t)K_ * 4;
  }
  lds[w][lane][0] = Q.a; lds[w][lane][1] = Q.b;
  lds[w][lane][2] = Q.c; lds[w][lane][3] = Q.d;
  __syncthreads();

  // exclusive sub-chunk prefix within this chunk (w is wave-uniform)
  m22 X = {1.f, 0.f, 0.f, 1.f};
  for (int ww = 0; ww < w; ++ww) {
    m22 q = {lds[ww][lane][0], lds[ww][lane][1], lds[ww][lane][2], lds[ww][lane][3]};
    X = mmul(q, X);
  }
  const float* ep = Cpre + (((size_t)(b * NT + chunk)) * K_ + chain) * 4;
  m22 E = {ep[0], ep[1], ep[2], ep[3]};
  m22 run = mmul(X, E);     // product of all M[t'] with t' < t0

  const float v0 = vvec[chain * 2 + 0];
  const float v1 = vvec[chain * 2 + 1];
#pragma unroll
  for (int j = 0; j < SUB; ++j) {
    run = mmul(mreg[j], run);                 // S[t0+j]
    float h0 = run.a * v0 + run.b * v1;
    float h1 = run.c * v0 + run.d * v1;
    *(unsigned*)&h[((size_t)(b * T_ + t0 + j)) * TWOK + 2 * chain] = pack2(h0, h1);
  }
}

// ---------------------------------------------------------------------------
// Kernel 3: out = x + out_scale * (h @ W_proj^T).  K = 512.
// A-tile (already bf16) staged with GLOBAL_LOAD_ASYNC_TO_LDS_B128: 16B/lane
// global->LDS DMA, tracked by ASYNCcnt, fenced with s_wait_asynccnt before
// the barrier that precedes the fragment ds_loads.
// ---------------------------------------------------------------------------
__global__ __launch_bounds__(256)
void k_gemm2(const unsigned short* __restrict__ h, const float* __restrict__ Wp,
             const float* __restrict__ x, const float* __restrict__ oscale,
             float* __restrict__ out) {
  __shared__ __align__(32) unsigned short lsA[64 * 32];
  __shared__ __align__(32) unsigned short lsB[128 * 32];

  const int tid   = threadIdx.x;
  const int lane  = tid & 31;
  const int wid   = tid >> 5;
  const int waveM = wid & 1;
  const int waveN = wid >> 1;
  const int rbase = blockIdx.x * 64;
  const int nbase = blockIdx.y * 128;

  const int ra = tid >> 2;            // 0..63   A row
  const int ca = (tid & 3) * 8;       // A col group (8 halfs = 16B)
  const int rb = tid >> 1;            // 0..127  B row
  const int cb = (tid & 1) * 16;      // B col group

  // LDS byte offset of this thread's A staging slot (ISA: low 32 bits of a
  // shared-aperture flat address are the LDS address).
  const unsigned ldsA_off = (unsigned)(size_t)&lsA[ra * 32 + ca];

  v8f acc[2][2] = {};

  for (int kt = 0; kt < TWOK / 32; ++kt) {
    const int kb = kt * 32;
    // stage A: async DMA 16B/lane from h (bf16) straight into LDS
    {
      unsigned long long gaddr =
          (unsigned long long)(size_t)(h + (size_t)(rbase + ra) * TWOK + kb + ca);
      asm volatile("global_load_async_to_lds_b128 %0, %1, off"
                   :: "v"(ldsA_off), "v"(gaddr) : "memory");
    }
    // stage B: W_proj rows -> bf16 (4 x b128 loads, 2 x b128 stores)
    {
      const float4* src = (const float4*)&Wp[(size_t)(nbase + rb) * TWOK + kb + cb];
      float4 f0 = src[0], f1 = src[1], f2 = src[2], f3 = src[3];
      uint4 p0, p1;
      p0.x = pack2(f0.x, f0.y); p0.y = pack2(f0.z, f0.w);
      p0.z = pack2(f1.x, f1.y); p0.w = pack2(f1.z, f1.w);
      p1.x = pack2(f2.x, f2.y); p1.y = pack2(f2.z, f2.w);
      p1.z = pack2(f3.x, f3.y); p1.w = pack2(f3.z, f3.w);
      *(uint4*)&lsB[rb * 32 + cb]     = p0;
      *(uint4*)&lsB[rb * 32 + cb + 8] = p1;
    }
    asm volatile("s_wait_asynccnt 0" ::: "memory");
    __syncthreads();

    const int klo = (lane >> 4) * 16;
    const int l15 = lane & 15;
    const int ar  = waveM * 32;
    const int bc  = waveN * 32;
    v16bf a0 = *(const v16bf*)&lsA[(ar + l15) * 32 + klo];
    v16bf a1 = *(const v16bf*)&lsA[(ar + 16 + l15) * 32 + klo];
    v16bf b0 = *(const v16bf*)&lsB[(bc + l15) * 32 + klo];
    v16bf b1 = *(const v16bf*)&lsB[(bc + 16 + l15) * 32 + klo];

    acc[0][0] = __builtin_amdgcn_wmma_f32_16x16x32_bf16(false, a0, false, b0, (short)0, acc[0][0], false, false);
    acc[0][1] = __builtin_amdgcn_wmma_f32_16x16x32_bf16(false, a0, false, b1, (short)0, acc[0][1], false, false);
    acc[1][0] = __builtin_amdgcn_wmma_f32_16x16x32_bf16(false, a1, false, b0, (short)0, acc[1][0], false, false);
    acc[1][1] = __builtin_amdgcn_wmma_f32_16x16x32_bf16(false, a1, false, b1, (short)0, acc[1][1], false, false);
    __syncthreads();
  }

  const int hi8 = (lane >> 4) * 8;
#pragma unroll
  for (int mi = 0; mi < 2; ++mi)
#pragma unroll
    for (int ni = 0; ni < 2; ++ni) {
#pragma unroll
      for (int j = 0; j < 8; ++j) {
        int row = rbase + waveM * 32 + mi * 16 + hi8 + j;
        int col = nbase + waveN * 32 + ni * 16 + (lane & 15);
        float v = acc[mi][ni][j];
        out[(size_t)row * D_ + col] =
            x[(size_t)row * D_ + col] + oscale[col] * v;
      }
    }
}

// ---------------------------------------------------------------------------
extern "C" void kernel_launch(void* const* d_in, const int* in_sizes, int n_in,
                              void* d_out, int out_size, void* d_ws, size_t ws_size,
                              hipStream_t stream) {
  const float* x   = (const float*)d_in[0];   // (B,T,D)
  const float* Wm  = (const float*)d_in[1];   // (4k, D)
  const float* vv  = (const float*)d_in[2];   // (k,2,1)
  const float* Wp  = (const float*)d_in[3];   // (D, 2k)
  const float* osc = (const float*)d_in[4];   // (D,)
  float* out = (float*)d_out;

  char* ws = (char*)d_ws;
  size_t off = 0;
  float* rrms = (float*)(ws + off);                 off += (size_t)RN * 4;                 // 32 KB
  unsigned short* Mbuf = (unsigned short*)(ws + off); off += (size_t)RN * FOURK * 2;       // 16 MB
  float* Ctot = (float*)(ws + off);                 off += (size_t)B_ * NT * K_ * 4 * 4;   // 256 KB
  float* Cpre = (float*)(ws + off);                 off += (size_t)B_ * NT * K_ * 4 * 4;   // 256 KB
  unsigned short* hbuf = (unsigned short*)(ws + off); off += (size_t)RN * TWOK * 2;        // 8 MB

  // 0) per-row inverse RMS
  k_rrms<<<RN / 8, 256, 0, stream>>>(x, rrms);

  // 1) M = rmsnorm(x) @ W_m^T, per-2x2 normalized, bf16
  dim3 g1(RN / 64, FOURK / 128);
  k_gemm1<<<g1, 256, 0, stream>>>(x, Wm, rrms, Mbuf);

  // 2) hierarchical prefix matmul scan over T, fused h = S @ v
  k_scan_tot  <<<B_ * 8 * NT, 256, 0, stream>>>(Mbuf, Ctot);
  k_scan_pre  <<<(B_ * K_) / 256, 256, 0, stream>>>(Ctot, Cpre);
  k_scan_apply<<<B_ * 8 * NT, 256, 0, stream>>>(Mbuf, Cpre, vv, hbuf);

  // 3) out = x + out_scale * (h @ W_proj^T)
  dim3 g3(RN / 64, D_ / 128);
  k_gemm2<<<g3, 256, 0, stream>>>(hbuf, Wp, x, osc, out);
}
